// GDNLayer_89455578841444
// MI455X (gfx1250) — matrix operand order repaired and checked by reading, not compiled, not run on hardware
//
#include <hip/hip_runtime.h>
#include <hip/hip_bf16.h>

// ---------------- problem constants ----------------
#define BB_   2
#define TT_   2048
#define CC_   1024
#define HH_   16
#define DD_   64
#define KK_   4
#define INNER_ 1024
#define MM_   (BB_ * TT_)          // 4096 rows

// ---------------- CDNA5 WMMA types ----------------
typedef __attribute__((ext_vector_type(16))) __bf16 v16bf;
typedef __attribute__((ext_vector_type(8)))  float  v8f;
typedef __attribute__((ext_vector_type(4)))  unsigned int u32x4;
typedef __attribute__((ext_vector_type(4)))  int v4i;

union Frag { u32x4 u[2]; v16bf v; };

__device__ __forceinline__ unsigned short f2bf(float f) {
  union { float f; unsigned u; } v; v.f = f;
  unsigned r = v.u + 0x7FFFu + ((v.u >> 16) & 1u);   // round-to-nearest-even
  return (unsigned short)(r >> 16);
}

// ---------------- async global->LDS staging (gfx1250 ASYNCcnt path) --------
// Probe-revealed signature: param0 = int4 addrspace(1)*, param1 = LDS side.
#if defined(__gfx1250__) && __has_builtin(__builtin_amdgcn_global_load_async_to_lds_b128)
#define HAVE_ASYNC_LDS 1
#define ASYNC_CP16(g, l)                                                     \
  __builtin_amdgcn_global_load_async_to_lds_b128(                            \
      (__attribute__((address_space(1))) v4i*)(g),                           \
      (__attribute__((address_space(3))) v4i*)(l), 0, 0)
#else
#define HAVE_ASYNC_LDS 0
#define ASYNC_CP16(g, l) (*(u32x4*)(l) = *(const u32x4*)(g))
#endif

__device__ __forceinline__ void async_wait_all() {
#if HAVE_ASYNC_LDS
#if __has_builtin(__builtin_amdgcn_s_wait_asynccnt)
  __builtin_amdgcn_s_wait_asynccnt(0);
#else
  asm volatile("s_wait_asynccnt 0x0" ::: "memory");
#endif
#endif
}

// ---------------- fp32 -> bf16 convert ----------------
__global__ __launch_bounds__(256)
void k_cvt_bf16(const float* __restrict__ in, unsigned short* __restrict__ out, int n) {
  int i = blockIdx.x * 256 + threadIdx.x;
  if (i < n) out[i] = f2bf(in[i]);
}

// W[O,C] fp32 (row-major) -> Wt[C, ldt] bf16 with Wt[c*ldt + o] = W[o*C + c]
__global__ __launch_bounds__(256)
void k_transp_bf16(const float* __restrict__ W, unsigned short* __restrict__ Wt,
                   int O, int C, int ldt) {
  int idx = blockIdx.x * 256 + threadIdx.x;
  if (idx < O * C) {
    int o = idx / C, c = idx % C;
    Wt[(size_t)c * ldt + o] = f2bf(W[(size_t)o * C + c]);
  }
}

// ---------------- LDS-staged double-buffered WMMA GEMM ----------------
// Y[M,N] = A[M,K] * Bt[K,N], bf16 in, f32 out.
// Block tile 128x128, K-step 32. 8 waves laid out 2x4; each wave owns a
// 64x32 sub-tile = 4x2 WMMA 16x16 tiles -> 8 v_wmma per K-step per wave.
#define GBM 128
#define GBN 128
#define GBK 32
#define A_STRIDE 40    // shorts/row: 80B, 16B aligned, bank-skewed
#define B_STRIDE 136   // shorts/row: 272B, 16B aligned
#define A_TILE (GBM * A_STRIDE)   // 5120 shorts
#define B_TILE (GBK * B_STRIDE)   // 4352 shorts

__device__ __forceinline__ void gemm_copy_tile(
    const unsigned short* __restrict__ A, const unsigned short* __restrict__ Bt,
    unsigned short* Ab, unsigned short* Bb,
    int mblk, int nblk, int k0, int Kdim, int Ndim, int tid) {
  // A tile: 128 rows x 32 cols, 64B/row -> 512 x 16B chunks, 2 issues/thread
  #pragma unroll
  for (int i = 0; i < 2; ++i) {
    int idx = tid + i * 256;
    int row = idx >> 2, c16 = idx & 3;
    const unsigned short* g = A + (size_t)(mblk + row) * Kdim + k0 + c16 * 8;
    unsigned short* l = Ab + row * A_STRIDE + c16 * 8;
    ASYNC_CP16(g, l);
  }
  // B tile: 32 rows x 128 cols, 256B/row -> 512 x 16B chunks
  #pragma unroll
  for (int i = 0; i < 2; ++i) {
    int idx = tid + i * 256;
    int row = idx >> 4, c16 = idx & 15;
    const unsigned short* g = Bt + (size_t)(k0 + row) * Ndim + nblk + c16 * 8;
    unsigned short* l = Bb + row * B_STRIDE + c16 * 8;
    ASYNC_CP16(g, l);
  }
}

__global__ __launch_bounds__(256)
void k_gemm_bf16_wmma(const unsigned short* __restrict__ A,
                      const unsigned short* __restrict__ Bt,
                      float* __restrict__ Y,
                      int Mdim, int Ndim, int Kdim, int ldc) {
  __shared__ __align__(16) unsigned short sm[2 * (A_TILE + B_TILE)];
  const int tid  = threadIdx.x;
  const int lane = tid & 31;
  const int wave = tid >> 5;
  const int wm   = wave >> 2;        // 0..1  -> 64-row half
  const int wn   = wave & 3;         // 0..3  -> 32-col slice
  const int mblk = blockIdx.y * GBM;
  const int nblk = blockIdx.x * GBN;

  // A fragment (ISA 16-bit A 16x32): lane -> M = lane&15;
  // lanes 0-15 carry K {0..7,16..23}, lanes 16-31 carry K {8..15,24..31}.
  const int mr = lane & 15;
  const int kb = (lane < 16) ? 0 : 8;

  v8f acc[4][2] = {};

  const int ntiles = Kdim / GBK;
  gemm_copy_tile(A, Bt, sm, sm + A_TILE, mblk, nblk, 0, Kdim, Ndim, tid);
  async_wait_all();
  __syncthreads();

  for (int kt = 0; kt < ntiles; ++kt) {
    if (kt + 1 < ntiles) {
      unsigned short* Abn = sm + ((kt + 1) & 1) * (A_TILE + B_TILE);
      gemm_copy_tile(A, Bt, Abn, Abn + A_TILE, mblk, nblk,
                     (kt + 1) * GBK, Kdim, Ndim, tid);
    }
    const unsigned short* Ab = sm + (kt & 1) * (A_TILE + B_TILE);
    const unsigned short* Bb = Ab + A_TILE;

    Frag bfr[2];
    #pragma unroll
    for (int tn = 0; tn < 2; ++tn) {        // B: lane -> N col = lane&15, K in A's order
      const unsigned short* bp = Bb + wn * 32 + tn * 16 + mr;
      #pragma unroll
      for (int s = 0; s < 16; ++s) {
        int kk = (s < 8) ? (kb + s) : (16 + kb + (s - 8));
        bfr[tn].v[s] = __builtin_bit_cast(__bf16, bp[kk * B_STRIDE]);
      }
    }
    #pragma unroll
    for (int tm = 0; tm < 4; ++tm) {
      Frag afr;
      const unsigned short* ap = Ab + (wm * 64 + tm * 16 + mr) * A_STRIDE + kb;
      afr.u[0] = *(const u32x4*)ap;
      afr.u[1] = *(const u32x4*)(ap + 16);
      #pragma unroll
      for (int tn = 0; tn < 2; ++tn)
        acc[tm][tn] = __builtin_amdgcn_wmma_f32_16x16x32_bf16(
            false, afr.v, false, bfr[tn].v, (short)0, acc[tm][tn], false, false);
    }
    async_wait_all();
    __syncthreads();
  }

  // C/D layout: VGPR r: lanes 0-15 -> M=r, lanes 16-31 -> M=8+r; N = lane&15.
  const int ncl  = lane & 15;
  const int madd = (lane >> 4) * 8;
  #pragma unroll
  for (int tm = 0; tm < 4; ++tm) {
    #pragma unroll
    for (int tn = 0; tn < 2; ++tn) {
      float* yp = Y + (size_t)(mblk + wm * 64 + tm * 16 + madd) * ldc
                    + nblk + wn * 32 + tn * 16 + ncl;
      #pragma unroll
      for (int r = 0; r < 8; ++r)
        yp[(size_t)r * ldc] = acc[tm][tn][r];
    }
  }
}

// ---------------- beta / decay heads ----------------
__global__ __launch_bounds__(256)
void k_ab(const float* __restrict__ x, const float* __restrict__ wa,
          const float* __restrict__ wb, const float* __restrict__ A_log,
          const float* __restrict__ dt_bias,
          float* __restrict__ beta, float* __restrict__ decay) {
  const int m = blockIdx.x;
  const int lane = threadIdx.x & 31;
  const int wv = threadIdx.x >> 5;          // 8 waves, 16 heads -> 2 per wave
  const float* xr = x + (size_t)m * CC_;
  for (int hh = wv; hh < HH_; hh += 8) {
    float sa = 0.f, sb = 0.f;
    for (int c = lane; c < CC_; c += 32) {
      float xv = xr[c];
      sa += xv * wa[(size_t)hh * CC_ + c];
      sb += xv * wb[(size_t)hh * CC_ + c];
    }
    #pragma unroll
    for (int msk = 16; msk > 0; msk >>= 1) {
      sa += __shfl_xor(sa, msk, 32);
      sb += __shfl_xor(sb, msk, 32);
    }
    if (lane == 0) {
      float a  = sa + dt_bias[hh];
      float sp = (a > 20.f) ? a : log1pf(expf(a));            // softplus
      decay[(size_t)m * HH_ + hh] = expf(-expf(A_log[hh]) * sp);
      beta [(size_t)m * HH_ + hh] = 1.f / (1.f + expf(-sb));
    }
  }
}

// ---------------- causal conv(K=4) + SiLU + L2 norm ----------------
// P is the fused qkvg projection output [M, 4096]: cols [q|k|v|g].
__global__ __launch_bounds__(64)
void k_conv_silu_norm(const float* __restrict__ P,
                      const float* __restrict__ wqc, const float* __restrict__ bqc,
                      const float* __restrict__ wkc, const float* __restrict__ bkc,
                      const float* __restrict__ wvc, const float* __restrict__ bvc,
                      float* __restrict__ qn, float* __restrict__ kn,
                      float* __restrict__ vn) {
  const int bt = blockIdx.x;            // b*T + t
  const int h  = blockIdx.y;
  const int d  = threadIdx.x;           // 0..63
  const int t  = bt % TT_;
  const int b  = bt / TT_;
  const int i  = h * DD_ + d;

  float aq = bqc[i], ak = bkc[i], av = bvc[i];
  #pragma unroll
  for (int j = 0; j < KK_; ++j) {
    int tt = t - (KK_ - 1) + j;
    if (tt >= 0) {
      size_t row = (size_t)(b * TT_ + tt) * 4096;
      aq += wqc[i * KK_ + j] * P[row + 0    + i];
      ak += wkc[i * KK_ + j] * P[row + 1024 + i];
      av += wvc[i * KK_ + j] * P[row + 2048 + i];
    }
  }
  float qy = aq / (1.f + expf(-aq));
  float ky = ak / (1.f + expf(-ak));
  float vy = av / (1.f + expf(-av));

  __shared__ float red[128];
  red[d] = qy * qy; red[64 + d] = ky * ky;
  __syncthreads();
  for (int s = 32; s > 0; s >>= 1) {
    if (d < s) { red[d] += red[d + s]; red[64 + d] += red[64 + d + s]; }
    __syncthreads();
  }
  float qs = rsqrtf(red[0]  + 1e-6f);    // l2_norm: sum + EPS
  float ks = rsqrtf(red[64] + 1e-6f);
  size_t o = (size_t)bt * INNER_ + i;
  qn[o] = qy * qs; kn[o] = ky * ks; vn[o] = vy;
}

// ---------------- sequential delta-rule scan ----------------
// One block per (b,h); thread e owns state column S[:,e] (64 regs, unrolled).
// Broadcast k/q/v through LDS as float4 to minimize per-step ds ops on the
// serial critical path.
__global__ __launch_bounds__(64)
void k_scan(const float* __restrict__ qn, const float* __restrict__ kn,
            const float* __restrict__ vn, const float* __restrict__ beta,
            const float* __restrict__ decay, float* __restrict__ osc) {
  const int b = blockIdx.x;
  const int h = blockIdx.y;
  const int e = threadIdx.x;
  __shared__ __align__(16) float sk[64], sq[64], sv[64];
  __shared__ float sbeta, sdecay;
  const float4* sk4 = (const float4*)sk;
  const float4* sq4 = (const float4*)sq;

  float S[64];
  #pragma unroll
  for (int d = 0; d < 64; ++d) S[d] = 0.f;

  for (int t = 0; t < TT_; ++t) {
    const size_t m = (size_t)b * TT_ + t;
    const size_t base = m * INNER_ + (size_t)h * DD_;
    sk[e] = kn[base + e];
    sq[e] = qn[base + e];
    sv[e] = vn[base + e];
    if (e == 0) { sbeta = beta[m * HH_ + h]; sdecay = decay[m * HH_ + h]; }
    __syncthreads();

    float kS = 0.f;
    #pragma unroll
    for (int d4 = 0; d4 < 16; ++d4) {
      float4 kv = sk4[d4];
      kS += kv.x * S[4 * d4 + 0] + kv.y * S[4 * d4 + 1]
          + kv.z * S[4 * d4 + 2] + kv.w * S[4 * d4 + 3];
    }
    const float wfac = sbeta * (sv[e] - kS);
    const float dec  = sdecay;
    float o = 0.f;
    #pragma unroll
    for (int d4 = 0; d4 < 16; ++d4) {
      float4 kv = sk4[d4];
      float4 qv = sq4[d4];
      S[4 * d4 + 0] = dec * S[4 * d4 + 0] + kv.x * wfac; o += qv.x * S[4 * d4 + 0];
      S[4 * d4 + 1] = dec * S[4 * d4 + 1] + kv.y * wfac; o += qv.y * S[4 * d4 + 1];
      S[4 * d4 + 2] = dec * S[4 * d4 + 2] + kv.z * wfac; o += qv.z * S[4 * d4 + 2];
      S[4 * d4 + 3] = dec * S[4 * d4 + 3] + kv.w * wfac; o += qv.w * S[4 * d4 + 3];
    }
    osc[base + e] = o;
    __syncthreads();
  }
}

// ---------------- RMS norm + SiLU gate -> bf16 ----------------
__global__ __launch_bounds__(64)
void k_gate(const float* __restrict__ osc, const float* __restrict__ P,
            const float* __restrict__ oscale, unsigned short* __restrict__ Ob) {
  const int bt = blockIdx.x;
  const int h  = blockIdx.y;
  const int d  = threadIdx.x;
  const int i  = h * DD_ + d;
  float ov = osc[(size_t)bt * INNER_ + i];
  __shared__ float red[64];
  red[d] = ov * ov;
  __syncthreads();
  for (int s = 32; s > 0; s >>= 1) {
    if (d < s) red[d] += red[d + s];
    __syncthreads();
  }
  float r = rsqrtf(red[0] / 64.f + 1e-6f);        // rms_norm: mean + EPS
  float g = P[(size_t)bt * 4096 + 3072 + i];
  float val = ov * r * oscale[d] * (g / (1.f + expf(-g)));
  Ob[(size_t)bt * INNER_ + i] = f2bf(val);
}

// ---------------- workspace layout (bytes) ----------------
#define OFF_XB    ((size_t)0)                      // x bf16         [M,C]      8 MB
#define OFF_WT    (OFF_XB   + (size_t)MM_*CC_*2)   // qkvg Wt bf16   [C,4096]   8 MB
#define OFF_P     (OFF_WT   + (size_t)CC_*4096*2)  // proj f32       [M,4096]  64 MB
#define OFF_QN    (OFF_P    + (size_t)MM_*4096*4)  // q normed f32   [M,1024]  16 MB
#define OFF_KN    (OFF_QN   + (size_t)MM_*INNER_*4)
#define OFF_VN    (OFF_KN   + (size_t)MM_*INNER_*4)
#define OFF_BETA  (OFF_VN   + (size_t)MM_*INNER_*4)
#define OFF_DEC   (OFF_BETA + (size_t)MM_*HH_*4)
#define OFF_OS    (OFF_DEC  + (size_t)MM_*HH_*4)   // scan out f32   [M,1024]
#define OFF_OB    (OFF_OS   + (size_t)MM_*INNER_*4)// gated bf16     [M,1024]
#define OFF_WOT   (OFF_OB   + (size_t)MM_*INNER_*2)// wo^T bf16      [1024,1024]

extern "C" void kernel_launch(void* const* d_in, const int* in_sizes, int n_in,
                              void* d_out, int out_size, void* d_ws, size_t ws_size,
                              hipStream_t stream) {
  (void)in_sizes; (void)n_in; (void)out_size; (void)ws_size;
  const float* x        = (const float*)d_in[0];
  const float* wq       = (const float*)d_in[1];
  const float* wk       = (const float*)d_in[2];
  const float* wv       = (const float*)d_in[3];
  const float* wg       = (const float*)d_in[4];
  const float* wo       = (const float*)d_in[5];
  const float* wa       = (const float*)d_in[6];
  const float* wb       = (const float*)d_in[7];
  const float* cqw      = (const float*)d_in[8];
  const float* cqb      = (const float*)d_in[9];
  const float* ckw      = (const float*)d_in[10];
  const float* ckb      = (const float*)d_in[11];
  const float* cvw      = (const float*)d_in[12];
  const float* cvb      = (const float*)d_in[13];
  const float* A_log    = (const float*)d_in[14];
  const float* dt_bias  = (const float*)d_in[15];
  const float* oscale   = (const float*)d_in[16];

  char* ws = (char*)d_ws;
  unsigned short* Xb   = (unsigned short*)(ws + OFF_XB);
  unsigned short* Wt   = (unsigned short*)(ws + OFF_WT);
  float*          P    = (float*)(ws + OFF_P);
  float*          qn   = (float*)(ws + OFF_QN);
  float*          kn   = (float*)(ws + OFF_KN);
  float*          vn   = (float*)(ws + OFF_VN);
  float*          beta = (float*)(ws + OFF_BETA);
  float*          dec  = (float*)(ws + OFF_DEC);
  float*          osc  = (float*)(ws + OFF_OS);
  unsigned short* Ob   = (unsigned short*)(ws + OFF_OB);
  unsigned short* Wot  = (unsigned short*)(ws + OFF_WOT);

  // 1) convert x to bf16
  k_cvt_bf16<<<(MM_ * CC_ + 255) / 256, 256, 0, stream>>>(x, Xb, MM_ * CC_);

  // 2) stack + transpose weights into [K,N] bf16
  const int we = INNER_ * CC_;
  k_transp_bf16<<<(we + 255) / 256, 256, 0, stream>>>(wq, Wt + 0,    INNER_, CC_, 4096);
  k_transp_bf16<<<(we + 255) / 256, 256, 0, stream>>>(wk, Wt + 1024, INNER_, CC_, 4096);
  k_transp_bf16<<<(we + 255) / 256, 256, 0, stream>>>(wv, Wt + 2048, INNER_, CC_, 4096);
  k_transp_bf16<<<(we + 255) / 256, 256, 0, stream>>>(wg, Wt + 3072, INNER_, CC_, 4096);
  k_transp_bf16<<<(we + 255) / 256, 256, 0, stream>>>(wo, Wot,       CC_, INNER_, CC_);

  // 3) fused qkvg projection: P[M,4096] = Xb[M,1024] * Wt[1024,4096]
  k_gemm_bf16_wmma<<<dim3(4096 / GBN, MM_ / GBM), 256, 0, stream>>>(
      Xb, Wt, P, MM_, 4096, CC_, 4096);

  // 4) beta / decay heads
  k_ab<<<MM_, 256, 0, stream>>>(x, wa, wb, A_log, dt_bias, beta, dec);

  // 5) conv + SiLU + L2 norm
  k_conv_silu_norm<<<dim3(MM_, HH_), 64, 0, stream>>>(
      P, cqw, cqb, ckw, ckb, cvw, cvb, qn, kn, vn);

  // 6) sequential scan over T
  k_scan<<<dim3(BB_, HH_), 64, 0, stream>>>(qn, kn, vn, beta, dec, osc);

  // 7) RMS norm + gate -> bf16
  k_gate<<<dim3(MM_, HH_), 64, 0, stream>>>(osc, P, oscale, Ob);

  // 8) output projection into d_out: [M,1024] = Ob[M,1024] * Wot[1024,1024]
  k_gemm_bf16_wmma<<<dim3(1024 / GBN, MM_ / GBM), 256, 0, stream>>>(
      Ob, Wot, (float*)d_out, MM_, CC_, INNER_, CC_);
}
